// MultiHeadAttention_28286654611720
// MI455X (gfx1250) — matrix operand order repaired and checked by reading, not compiled, not run on hardware
//
#include <hip/hip_runtime.h>
#include <hip/hip_bf16.h>

#define B_  4
#define L_  1024
#define E_  1024
#define H_  16
#define DH_ 64

typedef __attribute__((ext_vector_type(16))) __bf16 v16bf;
typedef __attribute__((ext_vector_type(8)))  __bf16 v8bf;
typedef __attribute__((ext_vector_type(8)))  float  v8f;
typedef __attribute__((ext_vector_type(4)))  float  v4f;

// ---------- fp32 -> bf16 helpers ----------
static __device__ inline v8bf cvt8(const float* __restrict__ p) {
  v4f a = *(const v4f*)p;
  v4f b = *(const v4f*)(p + 4);
  v8bf r;
  r[0] = (__bf16)a[0]; r[1] = (__bf16)a[1]; r[2] = (__bf16)a[2]; r[3] = (__bf16)a[3];
  r[4] = (__bf16)b[0]; r[5] = (__bf16)b[1]; r[6] = (__bf16)b[2]; r[7] = (__bf16)b[3];
  return r;
}

// One-time pack kernel: fp32 array -> bf16 array (n multiple of 8).
__global__ __launch_bounds__(256) void pack_bf16(const float* __restrict__ src,
                                                 __bf16* __restrict__ dst, int n) {
  const int i = (blockIdx.x * 256 + threadIdx.x) * 8;
  if (i < n) *(v8bf*)(dst + i) = cvt8(src + i);
}

// ---------- WMMA fragment loaders ----------
// A-matrix 16x32 bf16: lane half hf holds K = kk+[8*hf,+8) U kk+16+[8*hf,+8).
static __device__ inline v16bf a_frag_bf16(const __bf16* __restrict__ row, int kk, int hf) {
  v8bf lo = *(const v8bf*)(row + kk + hf * 8);
  v8bf hi = *(const v8bf*)(row + kk + 16 + hf * 8);
  v16bf f;
#pragma unroll
  for (int i = 0; i < 8; ++i) { f[i] = lo[i]; f[i + 8] = hi[i]; }
  return f;
}
static __device__ inline v16bf a_frag_f32(const float* __restrict__ row, int kk, int hf) {
  v8bf lo = cvt8(row + kk + hf * 8);
  v8bf hi = cvt8(row + kk + 16 + hf * 8);
  v16bf f;
#pragma unroll
  for (int i = 0; i < 8; ++i) { f[i] = lo[i]; f[i + 8] = hi[i]; }
  return f;
}
// B-matrix 32x16 bf16: lane = column; lane half hf holds K = kk+[8*hf,+8) U kk+16+[8*hf,+8).
static __device__ inline v16bf b_frag_bf16(const __bf16* __restrict__ row, int kk, int hf) {
  v8bf lo = *(const v8bf*)(row + kk + hf * 8);
  v8bf hi = *(const v8bf*)(row + kk + 16 + hf * 8);
  v16bf f;
#pragma unroll
  for (int i = 0; i < 8; ++i) { f[i] = lo[i]; f[i + 8] = hi[i]; }
  return f;
}

static __device__ inline v8f wmma_bf16(v16bf a, v16bf b, v8f c) {
  return __builtin_amdgcn_wmma_f32_16x16x32_bf16(
      /*neg_a=*/false, a, /*neg_b=*/false, b,
      /*c_mod=*/(short)0, c, /*reuse_a=*/false, /*reuse_b=*/false);
}

// ---------- projection GEMM: OUT = X @ W^T + bias (optionally *scale), all-bf16 operands ----------
// VMODE 0: out bf16, natural [B*L][E] layout.
// VMODE 1: out bf16, per-head transposed Vt[b][h][d][j].
template <int VMODE>
__global__ __launch_bounds__(256) void proj_kernel(
    const __bf16* __restrict__ X, const __bf16* __restrict__ W,
    const float* __restrict__ bias, float scale, __bf16* __restrict__ out) {
  const int lane = threadIdx.x & 31, wave = threadIdx.x >> 5;
  const int m = lane & 15, hf = lane >> 4;
  const int tileM = blockIdx.y * 128 + wave * 16;
  const int tileN = blockIdx.x * 64;
  const __bf16* arow = X + (size_t)(tileM + m) * E_;

  v8f acc[4] = {};
  for (int kk = 0; kk < E_; kk += 32) {
    __builtin_prefetch(arow + kk + 64, 0, 3);        // global_prefetch_b8, 2 k-steps ahead
    v16bf af = a_frag_bf16(arow, kk, hf);
#pragma unroll
    for (int n = 0; n < 4; ++n) {
      const __bf16* brow = W + (size_t)(tileN + 16 * n + m) * E_;
      acc[n] = wmma_bf16(af, b_frag_bf16(brow, kk, hf), acc[n]);
    }
  }
#pragma unroll
  for (int n = 0; n < 4; ++n) {
#pragma unroll
    for (int r = 0; r < 8; ++r) {
      const int i   = tileM + hf * 8 + r;            // output row (token)
      const int col = tileN + 16 * n + m;            // output col (embed)
      const float v = (acc[n][r] + bias[col]) * scale;
      if (VMODE == 0) {
        out[(size_t)i * E_ + col] = (__bf16)v;
      } else {  // V: store head-transposed Vt[b][h][d][j]
        const int bi = i >> 10, j = i & (L_ - 1);
        const int h = col >> 6, d = col & 63;
        out[((size_t)(bi * H_ + h) * DH_ + d) * L_ + j] = (__bf16)v;
      }
    }
  }
}

// ---------- scores = Q @ K^T per (b,h) -> weights slot of d_out (fp32) ----------
__global__ __launch_bounds__(256) void scores_kernel(
    const __bf16* __restrict__ qs, const __bf16* __restrict__ ks,
    float* __restrict__ wout) {
  const int lane = threadIdx.x & 31, wave = threadIdx.x >> 5;
  const int m = lane & 15, hf = lane >> 4;
  const int bh = blockIdx.z;
  const int bi = bh >> 4, h = bh & 15;
  const int tileM = blockIdx.y * 128 + wave * 16;
  const int tileN = blockIdx.x * 64;
  const __bf16* arow = qs + ((size_t)(bi * L_ + tileM + m) * E_ + h * DH_);

  v8f acc[4] = {};
#pragma unroll
  for (int kk = 0; kk < DH_; kk += 32) {
    v16bf af = a_frag_bf16(arow, kk, hf);
#pragma unroll
    for (int n = 0; n < 4; ++n) {
      const __bf16* brow = ks + ((size_t)(bi * L_ + tileN + 16 * n + m) * E_ + h * DH_);
      acc[n] = wmma_bf16(af, b_frag_bf16(brow, kk, hf), acc[n]);
    }
  }
#pragma unroll
  for (int n = 0; n < 4; ++n)
#pragma unroll
    for (int r = 0; r < 8; ++r) {
      const int i = tileM + hf * 8 + r;
      const int j = tileN + 16 * n + m;
      wout[((size_t)bh * L_ + i) * L_ + j] = acc[n][r];
    }
}

// ---------- row softmax, in place (used twice: attn and wrapper softmax) ----------
__global__ __launch_bounds__(256) void softmax_rows(float* __restrict__ p) {
  __shared__ float red[256];
  const int t = threadIdx.x;
  float* row = p + (size_t)blockIdx.x * L_;
  v4f x = *(const v4f*)(row + t * 4);

  float mx = fmaxf(fmaxf(x[0], x[1]), fmaxf(x[2], x[3]));
  red[t] = mx; __syncthreads();
  for (int s = 128; s > 0; s >>= 1) {
    if (t < s) red[t] = fmaxf(red[t], red[t + s]);
    __syncthreads();
  }
  mx = red[0]; __syncthreads();

  v4f e; float sum = 0.f;
#pragma unroll
  for (int i = 0; i < 4; ++i) { e[i] = __expf(x[i] - mx); sum += e[i]; }
  red[t] = sum; __syncthreads();
  for (int s = 128; s > 0; s >>= 1) {
    if (t < s) red[t] += red[t + s];
    __syncthreads();
  }
  const float inv = 1.0f / red[0];
#pragma unroll
  for (int i = 0; i < 4; ++i) e[i] *= inv;
  *(v4f*)(row + t * 4) = e;
}

// ---------- ctx = attn @ V per (b,h) -> ctx bf16, natural [B*L][E] ----------
// A-side streams 268MB of fp32 attn (not L2-resident) -> prefetch ahead.
__global__ __launch_bounds__(256) void ctx_kernel(
    const float* __restrict__ attn, const __bf16* __restrict__ vt,
    __bf16* __restrict__ ctx) {
  const int lane = threadIdx.x & 31, wave = threadIdx.x >> 5;
  const int m = lane & 15, hf = lane >> 4;
  const int bh = blockIdx.z;
  const int bi = bh >> 4, h = bh & 15;
  const int tileM = blockIdx.y * 128 + wave * 16;
  const float* arow = attn + ((size_t)bh * L_ + tileM + m) * L_;

  v8f acc[4] = {};
  for (int kk = 0; kk < L_; kk += 32) {
    __builtin_prefetch(arow + kk + 64, 0, 3);
    v16bf af = a_frag_f32(arow, kk, hf);
#pragma unroll
    for (int n = 0; n < 4; ++n) {
      const __bf16* brow = vt + ((size_t)bh * DH_ + 16 * n + m) * L_;
      acc[n] = wmma_bf16(af, b_frag_bf16(brow, kk, hf), acc[n]);
    }
  }
#pragma unroll
  for (int n = 0; n < 4; ++n)
#pragma unroll
    for (int r = 0; r < 8; ++r) {
      const int i = tileM + hf * 8 + r;
      const int d = 16 * n + m;
      ctx[(size_t)(bi * L_ + i) * E_ + h * DH_ + d] = (__bf16)acc[n][r];
    }
}

// ---------- out = ctx @ Wo^T + bo -> fp32 d_out ----------
__global__ __launch_bounds__(256) void outproj_kernel(
    const __bf16* __restrict__ ctx, const __bf16* __restrict__ Wo,
    const float* __restrict__ bo, float* __restrict__ out) {
  const int lane = threadIdx.x & 31, wave = threadIdx.x >> 5;
  const int m = lane & 15, hf = lane >> 4;
  const int tileM = blockIdx.y * 128 + wave * 16;
  const int tileN = blockIdx.x * 64;
  const __bf16* arow = ctx + (size_t)(tileM + m) * E_;

  v8f acc[4] = {};
  for (int kk = 0; kk < E_; kk += 32) {
    __builtin_prefetch(arow + kk + 64, 0, 3);
    v16bf af = a_frag_bf16(arow, kk, hf);
#pragma unroll
    for (int n = 0; n < 4; ++n) {
      const __bf16* brow = Wo + (size_t)(tileN + 16 * n + m) * E_;
      acc[n] = wmma_bf16(af, b_frag_bf16(brow, kk, hf), acc[n]);
    }
  }
#pragma unroll
  for (int n = 0; n < 4; ++n)
#pragma unroll
    for (int r = 0; r < 8; ++r) {
      const int i   = tileM + hf * 8 + r;
      const int col = tileN + 16 * n + m;
      out[(size_t)i * E_ + col] = acc[n][r] + bo[col];
    }
}

extern "C" void kernel_launch(void* const* d_in, const int* in_sizes, int n_in,
                              void* d_out, int out_size, void* d_ws, size_t ws_size,
                              hipStream_t stream) {
  const float* query = (const float*)d_in[0];
  const float* key_  = (const float*)d_in[1];
  const float* value = (const float*)d_in[2];
  const float* Wq = (const float*)d_in[3];
  const float* bq = (const float*)d_in[4];
  const float* Wk = (const float*)d_in[5];
  const float* bk = (const float*)d_in[6];
  const float* Wv = (const float*)d_in[7];
  const float* bv = (const float*)d_in[8];
  const float* Wo = (const float*)d_in[9];
  const float* bo = (const float*)d_in[10];

  float* out     = (float*)d_out;                      // [B, L, E]
  float* weights = out + (size_t)B_ * L_ * E_;         // [B, H, L, L]

  char* ws = (char*)d_ws;                              // 64 MB used
  __bf16* qs  = (__bf16*)(ws);                         // 8 MB  [B*L][E], pre-scaled
  __bf16* ks  = (__bf16*)(ws + ((size_t)8  << 20));    // 8 MB  [B*L][E]
  __bf16* vt  = (__bf16*)(ws + ((size_t)16 << 20));    // 8 MB  [B][H][Dh][L]
  __bf16* ctx = (__bf16*)(ws + ((size_t)24 << 20));    // 8 MB  [B*L][E]
  __bf16* xq  = (__bf16*)(ws + ((size_t)32 << 20));    // 8 MB  bf16 inputs
  __bf16* xk  = (__bf16*)(ws + ((size_t)40 << 20));    // 8 MB
  __bf16* xv  = (__bf16*)(ws + ((size_t)48 << 20));    // 8 MB
  __bf16* wqb = (__bf16*)(ws + ((size_t)56 << 20));    // 2 MB  bf16 weights
  __bf16* wkb = (__bf16*)(ws + ((size_t)58 << 20));    // 2 MB
  __bf16* wvb = (__bf16*)(ws + ((size_t)60 << 20));    // 2 MB
  __bf16* wob = (__bf16*)(ws + ((size_t)62 << 20));    // 2 MB

  const dim3 blk(256);
  const int nX = B_ * L_ * E_;   // 4,194,304
  const int nW = E_ * E_;        // 1,048,576

  // One-time bf16 packing of all GEMM operands.
  pack_bf16<<<dim3(nX / 2048), blk, 0, stream>>>(query, xq, nX);
  pack_bf16<<<dim3(nX / 2048), blk, 0, stream>>>(key_,  xk, nX);
  pack_bf16<<<dim3(nX / 2048), blk, 0, stream>>>(value, xv, nX);
  pack_bf16<<<dim3(nW / 2048), blk, 0, stream>>>(Wq, wqb, nW);
  pack_bf16<<<dim3(nW / 2048), blk, 0, stream>>>(Wk, wkb, nW);
  pack_bf16<<<dim3(nW / 2048), blk, 0, stream>>>(Wv, wvb, nW);
  pack_bf16<<<dim3(nW / 2048), blk, 0, stream>>>(Wo, wob, nW);

  const dim3 gproj(E_ / 64, (B_ * L_) / 128);          // 16 x 32
  const float scale = 0.125f;                          // Dh^-0.5 = 1/8

  proj_kernel<0><<<gproj, blk, 0, stream>>>(xq, wqb, bq, scale, qs);
  proj_kernel<0><<<gproj, blk, 0, stream>>>(xk, wkb, bk, 1.0f,  ks);
  proj_kernel<1><<<gproj, blk, 0, stream>>>(xv, wvb, bv, 1.0f,  vt);

  const dim3 gsc(L_ / 64, L_ / 128, B_ * H_);          // 16 x 8 x 64
  scores_kernel<<<gsc, blk, 0, stream>>>(qs, ks, weights);

  softmax_rows<<<dim3(B_ * H_ * L_), blk, 0, stream>>>(weights);   // attn, in place

  const dim3 gctx(1, L_ / 128, B_ * H_);               // 1 x 8 x 64
  ctx_kernel<<<gctx, blk, 0, stream>>>(weights, vt, ctx);

  outproj_kernel<<<gproj, blk, 0, stream>>>(ctx, wob, bo, out);

  softmax_rows<<<dim3(B_ * H_ * L_), blk, 0, stream>>>(weights);   // wrapper softmax
}
